// MoEActorCritic_24309514895613
// MI455X (gfx1250) — matrix operand order repaired and checked by reading, not compiled, not run on hardware
//
#include <hip/hip_runtime.h>
#include <hip/hip_bf16.h>

typedef __attribute__((ext_vector_type(16))) _Float16 v16h;
typedef __attribute__((ext_vector_type(8)))  _Float16 v8h;
typedef __attribute__((ext_vector_type(8)))  float    v8f;
typedef __attribute__((ext_vector_type(4)))  int      v4i;

#define NB 16384      // batch rows
#define NEXP 8        // experts
#define TOPK 2
#define NACT 32       // actions

#define AS1 __attribute__((address_space(1)))
#define AS3 __attribute__((address_space(3)))

#if __has_builtin(__builtin_amdgcn_global_load_async_to_lds_b128)
#define USE_ASYNC_LDS 1
#else
#define USE_ASYNC_LDS 0
#endif

__device__ inline void wait_async_lds() {
#if USE_ASYNC_LDS
#if __has_builtin(__builtin_amdgcn_s_wait_asynccnt)
  __builtin_amdgcn_s_wait_asynccnt(0);
#else
  asm volatile("s_wait_asynccnt 0x0" ::: "memory");
#endif
#endif
}

// ---------------- small utility kernels ----------------

__global__ void cvt_f32_f16(const float* __restrict__ src, _Float16* __restrict__ dst, int n) {
  for (int i = blockIdx.x * blockDim.x + threadIdx.x; i < n; i += gridDim.x * blockDim.x)
    dst[i] = (_Float16)src[i];
}

__global__ void zero_f32(float* __restrict__ p, int n) {
  for (int i = blockIdx.x * blockDim.x + threadIdx.x; i < n; i += gridDim.x * blockDim.x)
    p[i] = 0.0f;
}

__global__ void zero_i32(int* __restrict__ p, int n) {
  for (int i = blockIdx.x * blockDim.x + threadIdx.x; i < n; i += gridDim.x * blockDim.x)
    p[i] = 0;
}

// top-2 of gating logits; softmax+renorm collapses to a 2-way sigmoid.
__global__ void gate_top2(const float* __restrict__ logits,  // [NB, 8]
                          int* __restrict__ top_e, float* __restrict__ top_w,
                          int* __restrict__ counts) {
  int b = blockIdx.x * blockDim.x + threadIdx.x;
  if (b >= NB) return;
  float l[NEXP];
#pragma unroll
  for (int e = 0; e < NEXP; ++e) l[e] = logits[b * NEXP + e];
  int i1 = 0; float m1 = l[0];
#pragma unroll
  for (int e = 1; e < NEXP; ++e) if (l[e] > m1) { m1 = l[e]; i1 = e; }
  int i2 = -1; float m2 = -3.4e38f;
#pragma unroll
  for (int e = 0; e < NEXP; ++e) if (e != i1 && l[e] > m2) { m2 = l[e]; i2 = e; }
  float w1 = 1.0f / (1.0f + __expf(m2 - m1));
  float w2 = 1.0f - w1;
  top_e[2 * b + 0] = i1; top_e[2 * b + 1] = i2;
  top_w[2 * b + 0] = w1; top_w[2 * b + 1] = w2;
  atomicAdd(&counts[i1], 1);
  atomicAdd(&counts[i2], 1);
}

__global__ void prefix8(const int* __restrict__ counts, int* __restrict__ seg) {
  if (threadIdx.x == 0 && blockIdx.x == 0) {
    int acc = 0;
    for (int e = 0; e < NEXP; ++e) { seg[e] = acc; acc += counts[e]; }
    seg[NEXP] = acc;
  }
}

__global__ void place_slots(const int* __restrict__ top_e, const float* __restrict__ top_w,
                            const int* __restrict__ seg, int* __restrict__ cursor,
                            int* __restrict__ slot_row, float* __restrict__ slot_w) {
  int b = blockIdx.x * blockDim.x + threadIdx.x;
  if (b >= NB) return;
#pragma unroll
  for (int k = 0; k < TOPK; ++k) {
    int e = top_e[2 * b + k];
    int pos = atomicAdd(&cursor[e], 1);
    int s = seg[e] + pos;
    slot_row[s] = b;
    slot_w[s] = top_w[2 * b + k];
  }
}

// ---------------- WMMA GEMM ----------------
// C = epilogue(A[M,K] @ W[K,N] + bias)
// MODE 0: ELU -> f16 store (row = seg+gm)
// MODE 1: no act -> f32 store (gating logits)
// MODE 2: weighted atomic scatter-add into d_out (final expert layer)
// Block: 256 threads = 8 waves (4 x 2). Tile 128(M) x 128(N) x 32(K).
// Wave tile 32x64: 2 A-frags x 4 B-frags = 8 WMMA per K-step.
// Double-buffered LDS; global->LDS via async-to-LDS copies.

__device__ inline v8h zero8h() {
  v8h z;
#pragma unroll
  for (int i = 0; i < 8; ++i) z[i] = (_Float16)0.0f;
  return z;
}

template <int MODE, bool GATHER>
__global__ __launch_bounds__(256) void moe_gemm(
    const _Float16* __restrict__ A, const _Float16* __restrict__ W,
    const float* __restrict__ bias, void* __restrict__ Cout,
    int Mconst, const int* __restrict__ countPtr, const int* __restrict__ segPtr,
    const int* __restrict__ gatherIdx,                       // A row gather (layer E1)
    const int* __restrict__ scatRow, const float* __restrict__ scatW,  // MODE 2
    int K, int N, int lda, int ldc) {
  const int e = blockIdx.z;
  const int Mcnt = countPtr ? countPtr[e] : Mconst;
  const int m0 = blockIdx.y * 128;
  if (m0 >= Mcnt) return;                    // early-exit for sparse expert tiles
  const int seg = segPtr ? segPtr[e] : 0;
  W    += (size_t)e * K * N;
  bias += (size_t)e * N;

  const int n0blk = blockIdx.x * 128;
  const int tid = threadIdx.x;
  const int wave = tid >> 5, lane = tid & 31;
  const int wm = wave & 3, wn = wave >> 2;   // 4 x 2 wave grid; wave tile 32(M) x 64(N)
  const int g = lane >> 4, ln16 = lane & 15;

  __shared__ _Float16 As[2][128][40];        // 32 + 8 pad halfs, 80B row stride
  __shared__ _Float16 Bs[2][32][136];        // 128 + 8 pad halfs, 272B row stride

  // ---- loop-invariant staging setup (row indices hoisted out of K loop) ----
  const int rA = tid >> 2, cA = (tid & 3) << 3;    // A chunks: (rA, cA) and (rA+64, cA)
  const int rB = tid >> 4, cB = (tid & 15) << 3;   // B chunks: (rB, cB) and (rB+16, cB)
  const bool v0 = (m0 + rA) < Mcnt;
  const bool v1 = (m0 + rA + 64) < Mcnt;
  const bool vB = (n0blk + cB) < N;
  int ar0, ar1;
  if (GATHER) {
    ar0 = v0 ? gatherIdx[seg + m0 + rA] : 0;
    ar1 = v1 ? gatherIdx[seg + m0 + rA + 64] : 0;
  } else {
    ar0 = seg + m0 + rA;
    ar1 = seg + m0 + rA + 64;
  }
  const _Float16* pA0 = A + (size_t)ar0 * lda + cA;
  const _Float16* pA1 = A + (size_t)ar1 * lda + cA;
  const _Float16* pB0 = W + (size_t)rB * N + (n0blk + cB);
  const _Float16* pB1 = W + (size_t)(rB + 16) * N + (n0blk + cB);

  // zero-fill LDS slots owned by invalid lanes once (never overwritten later)
  if (!v0) { *(v8h*)&As[0][rA][cA] = zero8h(); *(v8h*)&As[1][rA][cA] = zero8h(); }
  if (!v1) { *(v8h*)&As[0][rA + 64][cA] = zero8h(); *(v8h*)&As[1][rA + 64][cA] = zero8h(); }
  if (!vB) {
    *(v8h*)&Bs[0][rB][cB] = zero8h();      *(v8h*)&Bs[1][rB][cB] = zero8h();
    *(v8h*)&Bs[0][rB + 16][cB] = zero8h(); *(v8h*)&Bs[1][rB + 16][cB] = zero8h();
  }

  auto stage = [&](int buf, int k0) {
#if USE_ASYNC_LDS
    if (v0) __builtin_amdgcn_global_load_async_to_lds_b128(
        (AS1 v4i*)(pA0 + k0), (AS3 v4i*)&As[buf][rA][cA], 0, 0);
    if (v1) __builtin_amdgcn_global_load_async_to_lds_b128(
        (AS1 v4i*)(pA1 + k0), (AS3 v4i*)&As[buf][rA + 64][cA], 0, 0);
    if (vB) {
      __builtin_amdgcn_global_load_async_to_lds_b128(
          (AS1 v4i*)(pB0 + (size_t)k0 * N), (AS3 v4i*)&Bs[buf][rB][cB], 0, 0);
      __builtin_amdgcn_global_load_async_to_lds_b128(
          (AS1 v4i*)(pB1 + (size_t)k0 * N), (AS3 v4i*)&Bs[buf][rB + 16][cB], 0, 0);
    }
#else
    if (v0) *(v8h*)&As[buf][rA][cA]      = *(const v8h*)(pA0 + k0);
    if (v1) *(v8h*)&As[buf][rA + 64][cA] = *(const v8h*)(pA1 + k0);
    if (vB) {
      *(v8h*)&Bs[buf][rB][cB]      = *(const v8h*)(pB0 + (size_t)k0 * N);
      *(v8h*)&Bs[buf][rB + 16][cB] = *(const v8h*)(pB1 + (size_t)k0 * N);
    }
#endif
  };

  v8f acc[2][4];
#pragma unroll
  for (int a = 0; a < 2; ++a)
#pragma unroll
    for (int b = 0; b < 4; ++b)
#pragma unroll
      for (int v = 0; v < 8; ++v) acc[a][b][v] = 0.0f;

  // prefetch first tile
  stage(0, 0);
  wait_async_lds();
  __syncthreads();

  int cur = 0;
  for (int k0 = 0; k0 < K; k0 += 32) {
    if (k0 + 32 < K) stage(cur ^ 1, k0 + 32);   // overlap copy of next tile

    // ---- fragments per ISA 7.12.2 layouts ----
    v16h afrag[2], bfrag[4];
#pragma unroll
    for (int mt = 0; mt < 2; ++mt) {
      const _Float16* p = &As[cur][wm * 32 + mt * 16 + ln16][g * 8];
      v8h lo = *(const v8h*)p;               // K = 8g .. 8g+7
      v8h hi = *(const v8h*)(p + 16);        // K = 16+8g .. 23+8g
#pragma unroll
      for (int j = 0; j < 8; ++j) { afrag[mt][j] = lo[j]; afrag[mt][8 + j] = hi[j]; }
    }
#pragma unroll
    for (int nt = 0; nt < 4; ++nt) {
      const _Float16* p = &Bs[cur][g * 8][wn * 64 + nt * 16 + ln16];  // column n = ln16, K = 8g+j / 16+8g+j
#pragma unroll
      for (int j = 0; j < 8; ++j) { bfrag[nt][j] = p[j * 136]; bfrag[nt][8 + j] = p[(16 + j) * 136]; }
    }
#pragma unroll
    for (int mt = 0; mt < 2; ++mt)
#pragma unroll
      for (int nt = 0; nt < 4; ++nt)
        acc[mt][nt] = __builtin_amdgcn_wmma_f32_16x16x32_f16(
            false, afrag[mt], false, bfrag[nt], (short)0, acc[mt][nt], false, false);

    wait_async_lds();      // next tile's copies landed in LDS
    __syncthreads();       // everyone done reading `cur`, next buffer visible
    cur ^= 1;
  }

  // ---- epilogue ----
#pragma unroll
  for (int mt = 0; mt < 2; ++mt)
#pragma unroll
    for (int nt = 0; nt < 4; ++nt)
#pragma unroll
      for (int v = 0; v < 8; ++v) {
        int gm = m0 + wm * 32 + mt * 16 + g * 8 + v;
        int gn = n0blk + wn * 64 + nt * 16 + ln16;
        if (gm < Mcnt && gn < N) {
          float val = acc[mt][nt][v] + bias[gn];
          if (MODE == 0) {
            val = val > 0.0f ? val : (__expf(val) - 1.0f);   // ELU
            ((_Float16*)Cout)[(size_t)(seg + gm) * ldc + gn] = (_Float16)val;
          } else if (MODE == 1) {
            ((float*)Cout)[(size_t)gm * ldc + gn] = val;
          } else {
            int s = seg + gm;
            atomicAdd(&((float*)Cout)[(size_t)scatRow[s] * ldc + gn], scatW[s] * val);
          }
        }
      }
}

// ---------------- host driver ----------------

static inline int cdiv_i(int a, int b) { return (a + b - 1) / b; }

extern "C" void kernel_launch(void* const* d_in, const int* in_sizes, int n_in,
                              void* d_out, int out_size, void* d_ws, size_t ws_size,
                              hipStream_t stream) {
  const float* obs = (const float*)d_in[0];
  const float* gw1 = (const float*)d_in[1];  const float* gb1 = (const float*)d_in[2];
  const float* gw2 = (const float*)d_in[3];  const float* gb2 = (const float*)d_in[4];
  const float* gw3 = (const float*)d_in[5];  const float* gb3 = (const float*)d_in[6];
  const float* ew1 = (const float*)d_in[7];  const float* eb1 = (const float*)d_in[8];
  const float* ew2 = (const float*)d_in[9];  const float* eb2 = (const float*)d_in[10];
  const float* ew3 = (const float*)d_in[11]; const float* eb3 = (const float*)d_in[12];
  const float* ew4 = (const float*)d_in[13]; const float* eb4 = (const float*)d_in[14];

  char* base = (char*)d_ws; size_t off = 0;
  auto alloc = [&](size_t bytes) -> void* {
    void* p = base + off; off = (off + bytes + 255) & ~(size_t)255; return p;
  };
  _Float16* obs_h  = (_Float16*)alloc((size_t)NB * 512 * 2);
  _Float16* gw1_h  = (_Float16*)alloc(512 * 256 * 2);
  _Float16* gw2_h  = (_Float16*)alloc(256 * 128 * 2);
  _Float16* gw3_h  = (_Float16*)alloc(128 * 8 * 2);
  _Float16* ew1_h  = (_Float16*)alloc((size_t)NEXP * 512 * 1024 * 2);
  _Float16* ew2_h  = (_Float16*)alloc((size_t)NEXP * 1024 * 512 * 2);
  _Float16* ew3_h  = (_Float16*)alloc((size_t)NEXP * 512 * 256 * 2);
  _Float16* ew4_h  = (_Float16*)alloc((size_t)NEXP * 256 * 32 * 2);
  _Float16* g1     = (_Float16*)alloc((size_t)NB * 256 * 2);
  _Float16* g2     = (_Float16*)alloc((size_t)NB * 128 * 2);
  float*    logits = (float*)   alloc((size_t)NB * NEXP * 4);
  int*      top_e  = (int*)     alloc((size_t)NB * TOPK * 4);
  float*    top_w  = (float*)   alloc((size_t)NB * TOPK * 4);
  int*      counts = (int*)     alloc(8 * 4);
  int*      cursor = (int*)     alloc(8 * 4);
  int*      seg    = (int*)     alloc(9 * 4);
  int*      slot_row = (int*)   alloc((size_t)NB * TOPK * 4);
  float*    slot_w   = (float*) alloc((size_t)NB * TOPK * 4);
  _Float16* h1 = (_Float16*)alloc((size_t)NB * TOPK * 1024 * 2);
  _Float16* h2 = (_Float16*)alloc((size_t)NB * TOPK * 512 * 2);
  _Float16* h3 = (_Float16*)alloc((size_t)NB * TOPK * 256 * 2);
  (void)ws_size; (void)n_in;

  auto cvt = [&](const float* s, _Float16* d, int n) {
    int blocks = cdiv_i(n, 256); if (blocks > 8192) blocks = 8192;
    cvt_f32_f16<<<blocks, 256, 0, stream>>>(s, d, n);
  };
  cvt(obs, obs_h, in_sizes[0]);
  cvt(gw1, gw1_h, in_sizes[1]); cvt(gw2, gw2_h, in_sizes[3]); cvt(gw3, gw3_h, in_sizes[5]);
  cvt(ew1, ew1_h, in_sizes[7]); cvt(ew2, ew2_h, in_sizes[9]);
  cvt(ew3, ew3_h, in_sizes[11]); cvt(ew4, ew4_h, in_sizes[13]);

  zero_i32<<<1, 64, 0, stream>>>(counts, 8);
  zero_i32<<<1, 64, 0, stream>>>(cursor, 8);
  zero_f32<<<cdiv_i(out_size, 256), 256, 0, stream>>>((float*)d_out, out_size);

  dim3 blk(256);
  // gating: 512 -> 256 -> 128 -> 8 (logits)
  moe_gemm<0, false><<<dim3(cdiv_i(256, 128), cdiv_i(NB, 128), 1), blk, 0, stream>>>(
      obs_h, gw1_h, gb1, g1, NB, nullptr, nullptr, nullptr, nullptr, nullptr,
      512, 256, 512, 256);
  moe_gemm<0, false><<<dim3(1, cdiv_i(NB, 128), 1), blk, 0, stream>>>(
      g1, gw2_h, gb2, g2, NB, nullptr, nullptr, nullptr, nullptr, nullptr,
      256, 128, 256, 128);
  moe_gemm<1, false><<<dim3(1, cdiv_i(NB, 128), 1), blk, 0, stream>>>(
      g2, gw3_h, gb3, logits, NB, nullptr, nullptr, nullptr, nullptr, nullptr,
      128, 8, 128, 8);

  // top-2 dispatch
  gate_top2<<<cdiv_i(NB, 256), 256, 0, stream>>>(logits, top_e, top_w, counts);
  prefix8<<<1, 32, 0, stream>>>(counts, seg);
  place_slots<<<cdiv_i(NB, 256), 256, 0, stream>>>(top_e, top_w, seg, cursor, slot_row, slot_w);

  // experts on compacted slots (gridDim.z = expert, early-exit past count)
  moe_gemm<0, true><<<dim3(1024 / 128, cdiv_i(NB, 128), NEXP), blk, 0, stream>>>(
      obs_h, ew1_h, eb1, h1, NB, counts, seg, slot_row, nullptr, nullptr,
      512, 1024, 512, 1024);
  moe_gemm<0, false><<<dim3(512 / 128, cdiv_i(NB, 128), NEXP), blk, 0, stream>>>(
      h1, ew2_h, eb2, h2, NB, counts, seg, nullptr, nullptr, nullptr,
      1024, 512, 1024, 512);
  moe_gemm<0, false><<<dim3(256 / 128, cdiv_i(NB, 128), NEXP), blk, 0, stream>>>(
      h2, ew3_h, eb3, h3, NB, counts, seg, nullptr, nullptr, nullptr,
      512, 256, 512, 256);
  moe_gemm<2, false><<<dim3(1, cdiv_i(NB, 128), NEXP), blk, 0, stream>>>(
      h3, ew4_h, eb4, d_out, NB, counts, seg, nullptr, slot_row, slot_w,
      256, 32, 256, 32);
}
